// GAT_16544214024770
// MI455X (gfx1250) — matrix operand order repaired and checked by reading, not compiled, not run on hardware
//
#include <hip/hip_runtime.h>
#include <hip/hip_bf16.h>

// ---------------------------------------------------------------------------
// 3-layer GAT for MI455X (gfx1250, wave32).
//   GEMMs      : v_wmma_f32_16x16x32_bf16 (bf16 in, f32 accum); A fragment
//                reloaded by each wave (one 16x16 tile per wave), B is L2-resident.
//   edge phase : streaming + hardware f32 atomics; ~1.8 GB gather/scatter per
//                layer vs 23.3 TB/s HBM => bandwidth bound. float4-vectorized
//                scatter computes each edge weight once per 4 channels.
//   L2 (192MB) holds the ~134MB inter-kernel working set.
// ---------------------------------------------------------------------------

typedef __bf16 bf16_t;
typedef __attribute__((ext_vector_type(16))) __bf16 v16bf;
typedef __attribute__((ext_vector_type(8)))  __bf16 v8bf;
typedef __attribute__((ext_vector_type(8)))  float  v8f;

#define NEG_SLOPE 0.2f

// ---- helpers ---------------------------------------------------------------

__device__ __forceinline__ unsigned short bf16_round(float f) {
  unsigned u = __float_as_uint(f);
  u += 0x7FFFu + ((u >> 16) & 1u);   // round-to-nearest-even
  return (unsigned short)(u >> 16);
}

// order-preserving encoding: enc(a) < enc(b) (unsigned) <=> a < b (float)
__device__ __forceinline__ unsigned fenc(float f) {
  unsigned u = __float_as_uint(f);
  return (u & 0x80000000u) ? ~u : (u | 0x80000000u);
}
__device__ __forceinline__ float fdec(unsigned u) {
  return __uint_as_float((u & 0x80000000u) ? (u & 0x7FFFFFFFu) : ~u);
}
#define ENC_NEG_INF 0x007FFFFFu   // fenc(-inf)

__device__ __forceinline__ float lrelu(float x) {
  return x > 0.f ? x : NEG_SLOPE * x;
}

// ---- elementwise / setup kernels (float4 vectorized) -----------------------

__global__ void k_f32_to_bf16_v4(const float4* __restrict__ x,
                                 uint2* __restrict__ y, long long n4) {
  long long i = (long long)blockIdx.x * blockDim.x + threadIdx.x;
  if (i >= n4) return;
  float4 v = x[i];
  uint2 o;
  o.x = (unsigned)bf16_round(v.x) | ((unsigned)bf16_round(v.y) << 16);
  o.y = (unsigned)bf16_round(v.z) | ((unsigned)bf16_round(v.w) << 16);
  y[i] = o;
}

// ELU then convert to bf16 for the next layer's GEMM input
__global__ void k_elu_bf16_v4(const float4* __restrict__ x,
                              uint2* __restrict__ y, long long n4) {
  long long i = (long long)blockIdx.x * blockDim.x + threadIdx.x;
  if (i >= n4) return;
  float4 v = x[i];
  v.x = v.x > 0.f ? v.x : (__expf(v.x) - 1.f);
  v.y = v.y > 0.f ? v.y : (__expf(v.y) - 1.f);
  v.z = v.z > 0.f ? v.z : (__expf(v.z) - 1.f);
  v.w = v.w > 0.f ? v.w : (__expf(v.w) - 1.f);
  uint2 o;
  o.x = (unsigned)bf16_round(v.x) | ((unsigned)bf16_round(v.y) << 16);
  o.y = (unsigned)bf16_round(v.z) | ((unsigned)bf16_round(v.w) << 16);
  y[i] = o;
}

// Wt[n, k] = bf16(W[k, n]); zero-pad n in [Nout, Noutp)
__global__ void k_convW(const float* __restrict__ W, unsigned short* __restrict__ Wt,
                        int K, int Nout, int Noutp) {
  int idx = blockIdx.x * blockDim.x + threadIdx.x;
  if (idx >= Noutp * K) return;
  int n = idx / K, k = idx - n * K;
  float v = (n < Nout) ? W[k * Nout + n] : 0.f;
  Wt[idx] = bf16_round(v);
}

__global__ void k_fill_u32_v4(uint4* __restrict__ p, unsigned val, long long n4) {
  long long i = (long long)blockIdx.x * blockDim.x + threadIdx.x;
  if (i >= n4) return;
  p[i] = make_uint4(val, val, val, val);
}

// ---- WMMA GEMM: feat[N, Noutp] = A(bf16 [N,128]) x Wt^T (bf16 [Noutp,128]) --
// One wave computes one 16x16 tile; K=128 unrolled into 4 v_wmma ops.

__global__ void __launch_bounds__(256)
k_gemm_wmma(const bf16_t* __restrict__ A, const bf16_t* __restrict__ Bt,
            float* __restrict__ C, int N, int Noutp) {
  const int K = 128;
  const int lane  = threadIdx.x & 31;
  const int wave  = threadIdx.x >> 5;
  const int nwav  = blockDim.x >> 5;
  const int tileM = blockIdx.x << 4;
  if (tileM >= N) return;

  const int m    = lane & 15;   // row (A) / col (B) within tile
  const int half = lane >> 4;   // 0: lanes 0-15, 1: lanes 16-31

  // A fragment: lane half==0 -> K {0..15}; half==1 -> K {16..31}
  const bf16_t* arow = A + (size_t)(tileM + m) * K;

  for (int tileN = wave << 4; tileN < Noutp; tileN += nwav << 4) {
    const bf16_t* brow = Bt + (size_t)(tileN + m) * K + half * 16;
    v8f acc = {};
#pragma unroll
    for (int k0 = 0; k0 < K; k0 += 32) {
      v16bf a = *(const v16bf*)(arow + k0 + half * 16);
      v16bf b = *(const v16bf*)(brow + k0);
      acc = __builtin_amdgcn_wmma_f32_16x16x32_bf16(
          false, a, false, b, (short)0, acc, false, false);
    }
    // D layout: VGPR v -> row v + 8*half, col = lane&15
    const int r0 = tileM + (half << 3);
    const int cn = tileN + m;
#pragma unroll
    for (int v = 0; v < 8; ++v)
      C[(size_t)(r0 + v) * Noutp + cn] = acc[v];
  }
}

// ---- attention scores: el/er [N,H] -----------------------------------------

__global__ void k_attn(const float* __restrict__ feat,
                       const float* __restrict__ al, const float* __restrict__ ar,
                       float* __restrict__ el, float* __restrict__ er,
                       int N, int H, int D, int ldf) {
  long long idx = (long long)blockIdx.x * blockDim.x + threadIdx.x;
  if (idx >= (long long)N * H) return;
  int n = (int)(idx / H), h = (int)(idx - (long long)n * H);
  const float* f = feat + (size_t)n * ldf + h * D;
  float sl = 0.f, sr = 0.f;
  for (int i = 0; i < D; ++i) { sl += f[i] * al[h * D + i]; sr += f[i] * ar[h * D + i]; }
  el[idx] = sl;
  er[idx] = sr;
}

// ---- edge softmax (3 streaming passes over edges) --------------------------
// One thread per edge; head loop unrolled (H is a template constant) so
// el[s*H..]/er[d*H..] become single b128 loads for H=4.

template <int H>
__global__ void k_edge_max(const int* __restrict__ src, const int* __restrict__ dst,
                           const float* __restrict__ el, const float* __restrict__ er,
                           unsigned* __restrict__ menc, long long E) {
  long long e = (long long)blockIdx.x * blockDim.x + threadIdx.x;
  if (e >= E) return;
  int s = src[e], d = dst[e];
#pragma unroll
  for (int h = 0; h < H; ++h) {
    float x = lrelu(el[(size_t)s * H + h] + er[(size_t)d * H + h]);
    atomicMax(&menc[(size_t)d * H + h], fenc(x));
  }
}

template <int H>
__global__ void k_edge_sum(const int* __restrict__ src, const int* __restrict__ dst,
                           const float* __restrict__ el, const float* __restrict__ er,
                           const unsigned* __restrict__ menc, float* __restrict__ denom,
                           long long E) {
  long long e = (long long)blockIdx.x * blockDim.x + threadIdx.x;
  if (e >= E) return;
  int s = src[e], d = dst[e];
#pragma unroll
  for (int h = 0; h < H; ++h) {
    float x = lrelu(el[(size_t)s * H + h] + er[(size_t)d * H + h]);
    float m = fdec(menc[(size_t)d * H + h]);
    atomicAdd(&denom[(size_t)d * H + h], __expf(x - m));
  }
}

// out[dst, c..c+3] += alpha(e,h) * feat[src, c..c+3].  One thread per
// (edge, 4-channel group): weight computed once per group, feat gathered b128.
template <int H, int D>
__global__ void k_edge_scatter_v4(const int* __restrict__ src, const int* __restrict__ dst,
                                  const float* __restrict__ el, const float* __restrict__ er,
                                  const unsigned* __restrict__ menc,
                                  const float* __restrict__ denom,
                                  const float* __restrict__ feat, float* __restrict__ out,
                                  long long E, int ldf, int ldo) {
  constexpr int G = (H * D) / 4;            // 4-channel groups per edge
  long long i = (long long)blockIdx.x * blockDim.x + threadIdx.x;
  if (i >= E * G) return;
  long long e = i / G;
  int c = (int)(i - e * G) * 4;             // channel base
  int h = c / D;
  int s = src[e], d = dst[e];
  size_t ih = (size_t)d * H + h;
  float x = lrelu(el[(size_t)s * H + h] + er[ih]);
  float w = __expf(x - fdec(menc[ih])) / denom[ih];
  float4 f = *(const float4*)(feat + (size_t)s * ldf + c);
  float* o = out + (size_t)d * ldo + c;
  atomicAdd(o + 0, f.x * w);
  atomicAdd(o + 1, f.y * w);
  atomicAdd(o + 2, f.z * w);
  atomicAdd(o + 3, f.w * w);
}

// ---- final log-softmax over 40 classes -------------------------------------

__global__ void k_log_softmax(const float* __restrict__ logits, float* __restrict__ out,
                              int N, int C) {
  int n = blockIdx.x * blockDim.x + threadIdx.x;
  if (n >= N) return;
  const float* x = logits + (size_t)n * C;
  float mx = x[0];
  for (int c = 1; c < C; ++c) mx = fmaxf(mx, x[c]);
  float s = 0.f;
  for (int c = 0; c < C; ++c) s += __expf(x[c] - mx);
  float ls = __logf(s);
  float* o = out + (size_t)n * C;
  for (int c = 0; c < C; ++c) o[c] = x[c] - mx - ls;
}

// ---------------------------------------------------------------------------
// Host orchestration
// ---------------------------------------------------------------------------

extern "C" void kernel_launch(void* const* d_in, const int* in_sizes, int n_in,
                              void* d_out, int out_size, void* d_ws, size_t ws_size,
                              hipStream_t stream) {
  const float* inputs = (const float*)d_in[0];
  const float* Wp[3]  = {(const float*)d_in[1], (const float*)d_in[4], (const float*)d_in[7]};
  const float* alp[3] = {(const float*)d_in[2], (const float*)d_in[5], (const float*)d_in[8]};
  const float* arp[3] = {(const float*)d_in[3], (const float*)d_in[6], (const float*)d_in[9]};
  const int* src = (const int*)d_in[10];
  const int* dst = (const int*)d_in[11];

  const int  N = in_sizes[0] / 128;          // 100000
  const long long E = in_sizes[10];          // 1,700,000
  const int  K = 128;

  // workspace carve-out (~134.5 MB total)
  char* p = (char*)d_ws;
  auto carve = [&](size_t bytes) -> void* {
    void* r = (void*)p;
    p += (bytes + 255) & ~(size_t)255;
    return r;
  };
  unsigned short* hbf   = (unsigned short*)carve((size_t)N * 128 * 2); // bf16 layer input
  unsigned short* Wt    = (unsigned short*)carve((size_t)128 * 128 * 2);
  float*          feat  = (float*)carve((size_t)N * 128 * 4);
  float*          outb  = (float*)carve((size_t)N * 128 * 4);
  float*          el    = (float*)carve((size_t)N * 4 * 4);
  float*          er    = (float*)carve((size_t)N * 4 * 4);
  unsigned*       menc  = (unsigned*)carve((size_t)N * 4 * 4);
  float*          denom = (float*)carve((size_t)N * 4 * 4);

  const int T = 256;
  auto nblk = [](long long n, int t) { return (unsigned)((n + t - 1) / t); };

  // input -> bf16 (N*128 divisible by 4)
  k_f32_to_bf16_v4<<<nblk((long long)N * 32, T), T, 0, stream>>>(
      (const float4*)inputs, (uint2*)hbf, (long long)N * 32);

  for (int L = 0; L < 3; ++L) {
    const int H     = (L == 2) ? 1 : 4;
    const int D     = (L == 2) ? 40 : 32;
    const int HD    = H * D;                 // out row stride (128 or 40)
    const int Noutp = (L == 2) ? 48 : 128;   // padded GEMM width (mult of 16)

    // W -> transposed bf16 [Noutp, K] (zero-padded columns)
    k_convW<<<nblk((long long)Noutp * K, T), T, 0, stream>>>(Wp[L], Wt, K, HD, Noutp);

    // feat = h @ W  (WMMA bf16 -> f32)
    k_gemm_wmma<<<nblk(N, 16), 256, 0, stream>>>(
        (const bf16_t*)hbf, (const bf16_t*)Wt, feat, N, Noutp);

    // attention scores
    k_attn<<<nblk((long long)N * H, T), T, 0, stream>>>(
        feat, alp[L], arp[L], el, er, N, H, D, Noutp);

    // init softmax accumulators + output (all counts divisible by 4)
    k_fill_u32_v4<<<nblk((long long)N * H / 4, T), T, 0, stream>>>(
        (uint4*)menc, ENC_NEG_INF, (long long)N * H / 4);
    k_fill_u32_v4<<<nblk((long long)N * H / 4, T), T, 0, stream>>>(
        (uint4*)denom, 0u, (long long)N * H / 4);
    k_fill_u32_v4<<<nblk((long long)N * HD / 4, T), T, 0, stream>>>(
        (uint4*)outb, 0u, (long long)N * HD / 4);

    // edge softmax: max, sum(exp), weighted scatter
    if (H == 4) {
      k_edge_max<4><<<nblk(E, T), T, 0, stream>>>(src, dst, el, er, menc, E);
      k_edge_sum<4><<<nblk(E, T), T, 0, stream>>>(src, dst, el, er, menc, denom, E);
      k_edge_scatter_v4<4, 32><<<nblk(E * 32, T), T, 0, stream>>>(
          src, dst, el, er, menc, denom, feat, outb, E, Noutp, HD);
    } else {
      k_edge_max<1><<<nblk(E, T), T, 0, stream>>>(src, dst, el, er, menc, E);
      k_edge_sum<1><<<nblk(E, T), T, 0, stream>>>(src, dst, el, er, menc, denom, E);
      k_edge_scatter_v4<1, 40><<<nblk(E * 10, T), T, 0, stream>>>(
          src, dst, el, er, menc, denom, feat, outb, E, Noutp, HD);
    }

    if (L < 2) {
      // ELU + bf16 for next layer (N*HD divisible by 4)
      k_elu_bf16_v4<<<nblk((long long)N * HD / 4, T), T, 0, stream>>>(
          (const float4*)outb, (uint2*)hbf, (long long)N * HD / 4);
    } else {
      // mean over H=1 is identity; final log-softmax over 40 classes
      k_log_softmax<<<nblk(N, T), T, 0, stream>>>(outb, (float*)d_out, N, HD);
    }
  }
}
